// Indexer_67353677136240
// MI455X (gfx1250) — matrix operand order repaired and checked by reading, not compiled, not run on hardware
//
#include <hip/hip_runtime.h>
#include <hip/hip_bf16.h>
#include <cstdint>

#define S_LEN 2048
#define DIMX  7168
#define QLR   1536
#define NH    64
#define DH    128
#define DR    64
#define TOPKN 512
#define QN    (NH * DH)    // 8192
#define PROJ_OUT (DH + NH) // 192
#define QTP   136          // padded LDS row stride (halves): 272B, 16B-aligned

typedef _Float16 half_t;
typedef __attribute__((ext_vector_type(16))) _Float16 v16h;
typedef __attribute__((ext_vector_type(8)))  _Float16 v8h;
typedef __attribute__((ext_vector_type(8)))  float    v8f;

__device__ __forceinline__ v8f vzero8() {
  v8f z;
#pragma unroll
  for (int i = 0; i < 8; ++i) z[i] = 0.0f;
  return z;
}

// A fragment (16x32 f16, M x K) from row-major global memory.
// Lane l<16: row M=l, K={0..7,16..23}; lane>=16: row M=l-16, K={8..15,24..31}.
__device__ __forceinline__ v16h load_frag_a(const half_t* __restrict__ base,
                                            int ld, int row0, int k0) {
  const int l  = threadIdx.x & 31;
  const int r  = row0 + (l & 15);
  const int ko = k0 + ((l >> 4) << 3);
  const v8h* p0 = reinterpret_cast<const v8h*>(base + (size_t)r * ld + ko);
  const v8h* p1 = reinterpret_cast<const v8h*>(base + (size_t)r * ld + ko + 16);
  v8h lo = *p0, hi = *p1;
  v16h a;
#pragma unroll
  for (int i = 0; i < 8; ++i) { a[i] = lo[i]; a[i + 8] = hi[i]; }
  return a;
}

// Same A-fragment layout but sourced from an LDS tile (padded stride ldp).
__device__ __forceinline__ v16h load_frag_a_lds(const half_t* base, int ldp,
                                                int k0) {
  const int l  = threadIdx.x & 31;
  const int r  = l & 15;
  const int ko = k0 + ((l >> 4) << 3);
  const v8h* p0 = reinterpret_cast<const v8h*>(base + r * ldp + ko);
  const v8h* p1 = reinterpret_cast<const v8h*>(base + r * ldp + ko + 16);
  v8h lo = *p0, hi = *p1;
  v16h a;
#pragma unroll
  for (int i = 0; i < 8; ++i) { a[i] = lo[i]; a[i + 8] = hi[i]; }
  return a;
}

// B fragment (32x16 f16, K x N) from row-major [N][K] memory.
// Lane l<16: col N=l, K={0..7,16..23}; lane>=16: K={8..15,24..31}.
__device__ __forceinline__ v16h load_frag_b(const half_t* __restrict__ base,
                                            int ld, int n0, int k0) {
  const int l  = threadIdx.x & 31;
  const int r  = n0 + (l & 15);
  const int ko = k0 + ((l >> 4) << 3);
  const v8h* p0 = reinterpret_cast<const v8h*>(base + (size_t)r * ld + ko);
  v8h lo = p0[0], hi = p0[2];
  v16h b;
#pragma unroll
  for (int i = 0; i < 8; ++i) { b[i] = lo[i]; b[i + 8] = hi[i]; }
  return b;
}

__device__ __forceinline__ v8f wmma_f16(v16h a, v16h b, v8f c) {
  return __builtin_amdgcn_wmma_f32_16x16x32_f16(false, a, false, b, (short)0, c,
                                                false, false);
}

// ---------------- Kernel 1: f32 -> f16 convert (grid-stride) ----------------
__global__ void cvt_f32_f16(const float* __restrict__ src,
                            half_t* __restrict__ dst, int n) {
  for (int i = blockIdx.x * blockDim.x + threadIdx.x; i < n;
       i += gridDim.x * blockDim.x)
    dst[i] = (half_t)src[i];
}

// ------- Kernel 2: pack wk (128x7168) + weights_proj (64x7168) -> [d][192] --
__global__ void pack_proj_weights(const float* __restrict__ wk,
                                  const float* __restrict__ wp,
                                  float* __restrict__ wt) {
  for (int i = blockIdx.x * blockDim.x + threadIdx.x; i < DIMX * PROJ_OUT;
       i += gridDim.x * blockDim.x) {
    int d = i / PROJ_OUT, j = i % PROJ_OUT;
    wt[i] = (j < DH) ? wk[(size_t)j * DIMX + d]
                     : wp[(size_t)(j - DH) * DIMX + d];
  }
}

// -- Kernel 3: per row t: k = WHT(rope(LN(x@wk^T)))/sqrt(128) (f16),
//              weights^T[h][t] = (x@wp^T) / (8*sqrt(128)) --------------------
__global__ void k_weights_kernel(const float* __restrict__ x,
                                 const float* __restrict__ wt,   // [d][192]
                                 const float* __restrict__ knw,
                                 const float* __restrict__ knb,
                                 const float* __restrict__ cosp,
                                 const float* __restrict__ sinp,
                                 half_t* __restrict__ k_out,     // [t][128]
                                 float* __restrict__ w_out) {    // [h][t]
  __shared__ float xrow[DIMX];
  __shared__ float kv[DH];
  __shared__ float wv[NH];
  __shared__ float red[2];
  const int t = blockIdx.x, tid = threadIdx.x;

  for (int i = tid; i < DIMX; i += blockDim.x) xrow[i] = x[(size_t)t * DIMX + i];
  __syncthreads();

  if (tid < PROJ_OUT) {
    float acc = 0.0f;
    for (int d = 0; d < DIMX; ++d)
      acc = fmaf(xrow[d], wt[(size_t)d * PROJ_OUT + tid], acc);
    if (tid < DH) kv[tid] = acc; else wv[tid - DH] = acc;
  }
  __syncthreads();

  if (tid == 0) {
    float m = 0.0f;
    for (int i = 0; i < DH; ++i) m += kv[i];
    m *= (1.0f / DH);
    float v = 0.0f;
    for (int i = 0; i < DH; ++i) { float d = kv[i] - m; v = fmaf(d, d, v); }
    v *= (1.0f / DH);
    red[0] = m;
    red[1] = rsqrtf(v + 1e-6f);
  }
  __syncthreads();
  if (tid < DH) kv[tid] = (kv[tid] - red[0]) * red[1] * knw[tid] + knb[tid];
  __syncthreads();

  if (tid < DR / 2) {  // non-interleaved RoPE on first 64 dims
    float c = cosp[t * (DR / 2) + tid], s = sinp[t * (DR / 2) + tid];
    float x1 = kv[tid], x2 = kv[tid + DR / 2];
    kv[tid]          = x1 * c - x2 * s;
    kv[tid + DR / 2] = x1 * s + x2 * c;
  }
  __syncthreads();

  for (int st = 1; st < DH; st <<= 1) {  // fast Walsh-Hadamard, 128 pts
    if (tid < DH / 2) {
      int i = ((tid / st) * st * 2) + (tid % st);
      float a = kv[i], b = kv[i + st];
      kv[i] = a + b; kv[i + st] = a - b;
    }
    __syncthreads();
  }
  const float inv_sqrt_dh = 0.08838834764831845f;  // 128^-0.5
  if (tid < DH) k_out[(size_t)t * DH + tid] = (half_t)(kv[tid] * inv_sqrt_dh);
  if (tid < NH)
    w_out[(size_t)tid * S_LEN + t] = wv[tid] * (0.125f * inv_sqrt_dh);
}

// ---- Kernel 4: q = qr_h(2048x1536) @ wqb_h^T(1536x8192), WMMA f16->f32 -----
__global__ void qproj_gemm(const half_t* __restrict__ A,   // [2048][1536]
                           const half_t* __restrict__ Bw,  // [8192][1536]
                           half_t* __restrict__ C) {       // [2048][8192]
  const int wave = blockIdx.x * (blockDim.x >> 5) + (threadIdx.x >> 5);
  const int tn = wave & 255, tm = wave >> 8;   // 64 x 256 tiles of 32x32
  const int m0 = tm * 32, n0 = tn * 32;
  v8f c00 = vzero8(), c01 = vzero8(), c10 = vzero8(), c11 = vzero8();
  for (int k0 = 0; k0 < QLR; k0 += 32) {
    v16h a0 = load_frag_a(A, QLR, m0,      k0);
    v16h a1 = load_frag_a(A, QLR, m0 + 16, k0);
    v16h b0 = load_frag_b(Bw, QLR, n0,      k0);
    v16h b1 = load_frag_b(Bw, QLR, n0 + 16, k0);
    c00 = wmma_f16(a0, b0, c00);
    c01 = wmma_f16(a0, b1, c01);
    c10 = wmma_f16(a1, b0, c10);
    c11 = wmma_f16(a1, b1, c11);
  }
  const int l = threadIdx.x & 31;
  const int col = l & 15, rb = (l >> 4) << 3;
#pragma unroll
  for (int i = 0; i < 8; ++i) {
    size_t r0 = (size_t)(m0 + rb + i) * QN;
    size_t r1 = (size_t)(m0 + 16 + rb + i) * QN;
    C[r0 + n0 + col]      = (half_t)c00[i];
    C[r0 + n0 + 16 + col] = (half_t)c01[i];
    C[r1 + n0 + col]      = (half_t)c10[i];
    C[r1 + n0 + 16 + col] = (half_t)c11[i];
  }
}

// ---- Kernel 5: per (t,h): in-place RoPE (first 64 dims) + WHT + /sqrt(128) --
__global__ void q_post_kernel(half_t* __restrict__ Q,
                              const float* __restrict__ cosp,
                              const float* __restrict__ sinp) {
  __shared__ float qv[DH];
  const int t = blockIdx.x >> 6, h = blockIdx.x & 63, tid = threadIdx.x;
  half_t* base = Q + (size_t)t * QN + h * DH;
  if (tid < DH) qv[tid] = (float)base[tid];
  __syncthreads();
  if (tid < DR / 2) {
    float c = cosp[t * (DR / 2) + tid], s = sinp[t * (DR / 2) + tid];
    float x1 = qv[tid], x2 = qv[tid + DR / 2];
    qv[tid]          = x1 * c - x2 * s;
    qv[tid + DR / 2] = x1 * s + x2 * c;
  }
  __syncthreads();
  for (int st = 1; st < DH; st <<= 1) {
    if (tid < DH / 2) {
      int i = ((tid / st) * st * 2) + (tid % st);
      float a = qv[i], b = qv[i + st];
      qv[i] = a + b; qv[i + st] = a - b;
    }
    __syncthreads();
  }
  if (tid < DH) base[tid] = (half_t)(qv[tid] * 0.08838834764831845f);
}

// ---- Kernel 6: index_score[t][s] = sum_h relu(q[t,h,:].k[s,:]) * w[t,h] + mask
// Block (8 waves) owns one 16-row t-tile and 256 s-columns; per head the
// 16x128 q slab is staged once into LDS (b128 per thread) and each wave reads
// its A fragments with ds_load_b128. K fragments live in registers for the
// whole h loop. Weights are read transposed [h][t] as two b128 per lane.
__global__ void score_kernel(const half_t* __restrict__ Q,   // [2048][8192]
                             const half_t* __restrict__ Kk,  // [2048][128]
                             const float* __restrict__ Wt,   // [64][2048]
                             const float* __restrict__ mask, // [2048][2048]
                             float* __restrict__ out) {      // [2048][2048]
  __shared__ alignas(16) half_t qtile[16 * QTP];

  const int tt = blockIdx.x >> 3;          // 128 t-tiles
  const int sblk = blockIdx.x & 7;         // 8 s-blocks of 256 columns
  const int waveId = threadIdx.x >> 5;     // 8 waves per block
  const int t0 = tt * 16;
  const int s0 = sblk * 256 + waveId * 32;

  v16h bfr[2][4];
#pragma unroll
  for (int st = 0; st < 2; ++st)
#pragma unroll
    for (int ks = 0; ks < 4; ++ks)
      bfr[st][ks] = load_frag_b(Kk, DH, s0 + st * 16, ks * 32);

  v8f acc0 = vzero8(), acc1 = vzero8();
  const int l = threadIdx.x & 31;
  const int col = l & 15, rb = (l >> 4) << 3;

  const int srow = threadIdx.x >> 4;            // staging: 16B per thread
  const int scol = (threadIdx.x & 15) << 3;

  for (int h = 0; h < NH; ++h) {
    // stage q[t0..t0+15][h*128..h*128+127] into LDS (padded rows)
    const v8h* gsrc = reinterpret_cast<const v8h*>(
        Q + (size_t)(t0 + srow) * QN + h * DH + scol);
    *reinterpret_cast<v8h*>(&qtile[srow * QTP + scol]) = *gsrc;
    __syncthreads();

    v16h a[4];
#pragma unroll
    for (int ks = 0; ks < 4; ++ks) a[ks] = load_frag_a_lds(qtile, QTP, ks * 32);

    v8f qk0 = vzero8(), qk1 = vzero8();
#pragma unroll
    for (int ks = 0; ks < 4; ++ks) qk0 = wmma_f16(a[ks], bfr[0][ks], qk0);
#pragma unroll
    for (int ks = 0; ks < 4; ++ks) qk1 = wmma_f16(a[ks], bfr[1][ks], qk1);

    float wv_[8];
    const float4* w4 =
        reinterpret_cast<const float4*>(Wt + (size_t)h * S_LEN + t0 + rb);
    *reinterpret_cast<float4*>(&wv_[0]) = w4[0];
    *reinterpret_cast<float4*>(&wv_[4]) = w4[1];
#pragma unroll
    for (int i = 0; i < 8; ++i) {
      acc0[i] = fmaf(fmaxf(qk0[i], 0.0f), wv_[i], acc0[i]);
      acc1[i] = fmaf(fmaxf(qk1[i], 0.0f), wv_[i], acc1[i]);
    }
    __syncthreads();  // protect qtile before next head overwrites it
  }

#pragma unroll
  for (int i = 0; i < 8; ++i) {
    int t = t0 + rb + i;
    int sa = s0 + col, sb = s0 + 16 + col;
    out[(size_t)t * S_LEN + sa] = acc0[i] + mask[(size_t)t * S_LEN + sa];
    out[(size_t)t * S_LEN + sb] = acc1[i] + mask[(size_t)t * S_LEN + sb];
  }
}

// ---- Kernel 7: per-row top-512 of 2048 via bitonic sort in LDS --------------
__global__ void topk_kernel(const float* __restrict__ score,
                            int* __restrict__ idx_out) {
  __shared__ float v[S_LEN];
  __shared__ int   id[S_LEN];
  const int row = blockIdx.x, tid = threadIdx.x;
  for (int i = tid; i < S_LEN; i += blockDim.x) {
    v[i] = score[(size_t)row * S_LEN + i];
    id[i] = i;
  }
  __syncthreads();
  for (int k = 2; k <= S_LEN; k <<= 1) {
    for (int j = k >> 1; j > 0; j >>= 1) {
      for (int i = tid; i < S_LEN; i += blockDim.x) {
        int ixj = i ^ j;
        if (ixj > i) {
          bool desc = ((i & k) == 0);  // descending overall (inverted bitonic)
          float va = v[i], vb = v[ixj];
          int ia = id[i], ib = id[ixj];
          bool a_first = (va > vb) || (va == vb && ia < ib);
          bool do_swap = desc ? !a_first : a_first;
          if (do_swap) { v[i] = vb; v[ixj] = va; id[i] = ib; id[ixj] = ia; }
        }
      }
      __syncthreads();
    }
  }
  for (int i = tid; i < TOPKN; i += blockDim.x)
    idx_out[(size_t)row * TOPKN + i] = id[i];
}

// ----------------------------------------------------------------------------
extern "C" void kernel_launch(void* const* d_in, const int* in_sizes, int n_in,
                              void* d_out, int out_size, void* d_ws,
                              size_t ws_size, hipStream_t stream) {
  const float* x    = (const float*)d_in[0];   // [1,2048,7168]
  const float* qr   = (const float*)d_in[1];   // [1,2048,1536]
  const float* wq_b = (const float*)d_in[2];   // [8192,1536]
  const float* wk   = (const float*)d_in[3];   // [128,7168]
  const float* wp   = (const float*)d_in[4];   // [64,7168]
  const float* knw  = (const float*)d_in[5];   // [128]
  const float* knb  = (const float*)d_in[6];   // [128]
  const float* cosp = (const float*)d_in[7];   // [2048,32]
  const float* sinp = (const float*)d_in[8];   // [2048,32]
  const float* mask = (const float*)d_in[9];   // [2048,2048]
  (void)in_sizes; (void)n_in; (void)out_size; (void)ws_size;

  // workspace carve-up (all 256B aligned)
  char* ws = (char*)d_ws;
  half_t* qr_h  = (half_t*)(ws);                             // 6,291,456 B
  half_t* wqb_h = (half_t*)(ws + 6291456);                   // 25,165,824 B
  half_t* q_h   = (half_t*)(ws + 6291456 + 25165824);        // 33,554,432 B
  half_t* k_h   = (half_t*)(ws + 65011712);                  //    524,288 B
  float*  w_f   = (float*) (ws + 65011712 + 524288);         //    524,288 B
  float*  wkt   = (float*) (ws + 65011712 + 1048576);        //  5,505,024 B

  int* idx_out     = (int*)d_out;
  float* score_out = (float*)d_out + (size_t)S_LEN * TOPKN;

  cvt_f32_f16<<<4096, 256, 0, stream>>>(qr, qr_h, S_LEN * QLR);
  cvt_f32_f16<<<8192, 256, 0, stream>>>(wq_b, wqb_h, QN * QLR);
  pack_proj_weights<<<2048, 256, 0, stream>>>(wk, wp, wkt);
  k_weights_kernel<<<S_LEN, 256, 0, stream>>>(x, wkt, knw, knb, cosp, sinp,
                                              k_h, w_f);
  // 64x256 tiles of 32x32 -> 16384 waves -> 2048 blocks of 8 waves
  qproj_gemm<<<2048, 256, 0, stream>>>(qr_h, wqb_h, q_h);
  q_post_kernel<<<S_LEN * NH, 128, 0, stream>>>(q_h, cosp, sinp);
  // 128 t-tiles x 8 s-blocks -> 1024 blocks; block = one t-tile, 256 s cols
  score_kernel<<<1024, 256, 0, stream>>>(q_h, k_h, w_f, mask, score_out);
  topk_kernel<<<S_LEN, 256, 0, stream>>>(score_out, idx_out);
}
